// AttnDecoder_89893665505788
// MI455X (gfx1250) — matrix operand order repaired and checked by reading, not compiled, not run on hardware
//
#include <hip/hip_runtime.h>
#include <hip/hip_bf16.h>
#include <cstdint>

// Sizes fixed by the reference
#define B_DIM  1024
#define T_DIM  64
#define CH_DIM 512
#define H_DIM  512

typedef __attribute__((ext_vector_type(16))) __bf16 v16bf;
typedef __attribute__((ext_vector_type(8)))  float  v8f;

__device__ __forceinline__ float sigm(float x) { return 1.0f / (1.0f + __expf(-x)); }

// ---------------------------------------------------------------------------
// f32 -> bf16 elementwise convert
// ---------------------------------------------------------------------------
__global__ void cvt_bf16_kernel(const float* __restrict__ in, __bf16* __restrict__ out, size_t n) {
    size_t i = (size_t)blockIdx.x * blockDim.x + threadIdx.x;
    if (i < n) out[i] = (__bf16)in[i];
}

// zci[b,k] = b2[k] + bci[k] + sum_t cis[b,t] * Wci[k,t]
__global__ void zci_kernel(const float* __restrict__ cis, const float* __restrict__ Wci,
                           const float* __restrict__ bci, const float* __restrict__ b2,
                           float* __restrict__ zci) {
    int idx = blockIdx.x * blockDim.x + threadIdx.x;   // B*CH
    int b = idx >> 9, k = idx & 511;
    const float* crow = cis + (size_t)b * T_DIM;
    const float* wrow = Wci + (size_t)k * T_DIM;
    float acc = bci[k] + b2[k];
#pragma unroll 8
    for (int t = 0; t < T_DIM; ++t) acc += crow[t] * wrow[t];
    zci[idx] = acc;
}

// bvec[i] = bih[i] + bhh[i]   (length 4H)
__global__ void bvec_kernel(const float* __restrict__ bih, const float* __restrict__ bhh,
                            float* __restrict__ bvec) {
    int i = blockIdx.x * blockDim.x + threadIdx.x;
    bvec[i] = bih[i] + bhh[i];
}

// ---------------------------------------------------------------------------
// WMMA bf16 GEMM:  out[M,N] = A[M,K] @ W[N,K]^T (+ biasN[n]) (+ biasG[m>>gshift, n])
// A row-major with leading dim lda, W row-major (N,K) with leading dim ldw.
// Wave tile 32x64 (2 A-frags x 4 B-frags -> 8 wmma per K-chunk of 32).
// Block = 256 threads = 8 waves -> 256 rows x 64 cols per block.
// grid = (N/64, M/256). Output f32 (outF) or bf16 (outB != nullptr).
// ---------------------------------------------------------------------------
__device__ __forceinline__ v16bf load_a_frag(const __bf16* __restrict__ A, int lda,
                                             int row, int kk, int lane) {
    // ISA 16-bit A 16x32 layout: lanes 0-15: M=lane, K={0..7,16..23};
    //                            lanes 16-31: M=lane-16, K={8..15,24..31}
    const int hi = lane >> 4;
    const int m  = row + (lane & 15);
    const __bf16* p = A + (size_t)m * lda + kk + hi * 8;
    v16bf a;
    *reinterpret_cast<uint4*>(&a)       = *reinterpret_cast<const uint4*>(p);
    *(reinterpret_cast<uint4*>(&a) + 1) = *reinterpret_cast<const uint4*>(p + 16);
    return a;
}

__device__ __forceinline__ v16bf load_b_frag(const __bf16* __restrict__ W, int ldw,
                                             int col, int kk, int lane) {
    // ISA 16-bit B 32x16 layout: lanes 0-15: N=lane, K={0..7,16..23};
    //                            lanes 16-31: N=lane-16, K={8..15,24..31}
    const int hi = lane >> 4;
    const int n  = col + (lane & 15);
    const __bf16* p = W + (size_t)n * ldw + kk + hi * 8;
    v16bf b;
    *reinterpret_cast<uint4*>(&b)       = *reinterpret_cast<const uint4*>(p);
    *(reinterpret_cast<uint4*>(&b) + 1) = *reinterpret_cast<const uint4*>(p + 16);
    return b;
}

__global__ void gemm_wmma_kernel(const __bf16* __restrict__ A, int lda,
                                 const __bf16* __restrict__ W, int ldw,
                                 float* __restrict__ outF, __bf16* __restrict__ outB,
                                 const float* __restrict__ biasN,
                                 const float* __restrict__ biasG, int gshift,
                                 int N, int K) {
    const int lane = threadIdx.x & 31;
    const int wave = threadIdx.x >> 5;
    const int col  = blockIdx.x * 64;
    const int row  = blockIdx.y * 256 + wave * 32;

    v8f c[2][4];
#pragma unroll
    for (int i = 0; i < 2; ++i)
#pragma unroll
        for (int j = 0; j < 4; ++j) c[i][j] = {};

    for (int kk = 0; kk < K; kk += 32) {
        v16bf a0 = load_a_frag(A, lda, row,      kk, lane);
        v16bf a1 = load_a_frag(A, lda, row + 16, kk, lane);
        v16bf b0 = load_b_frag(W, ldw, col,      kk, lane);
        v16bf b1 = load_b_frag(W, ldw, col + 16, kk, lane);
        v16bf b2 = load_b_frag(W, ldw, col + 32, kk, lane);
        v16bf b3 = load_b_frag(W, ldw, col + 48, kk, lane);
        c[0][0] = __builtin_amdgcn_wmma_f32_16x16x32_bf16(false, a0, false, b0, (short)0, c[0][0], false, false);
        c[0][1] = __builtin_amdgcn_wmma_f32_16x16x32_bf16(false, a0, false, b1, (short)0, c[0][1], false, false);
        c[0][2] = __builtin_amdgcn_wmma_f32_16x16x32_bf16(false, a0, false, b2, (short)0, c[0][2], false, false);
        c[0][3] = __builtin_amdgcn_wmma_f32_16x16x32_bf16(false, a0, false, b3, (short)0, c[0][3], false, false);
        c[1][0] = __builtin_amdgcn_wmma_f32_16x16x32_bf16(false, a1, false, b0, (short)0, c[1][0], false, false);
        c[1][1] = __builtin_amdgcn_wmma_f32_16x16x32_bf16(false, a1, false, b1, (short)0, c[1][1], false, false);
        c[1][2] = __builtin_amdgcn_wmma_f32_16x16x32_bf16(false, a1, false, b2, (short)0, c[1][2], false, false);
        c[1][3] = __builtin_amdgcn_wmma_f32_16x16x32_bf16(false, a1, false, b3, (short)0, c[1][3], false, false);
    }

    // C/D layout: VGPR r -> lanes 0-15: M=r, N=lane ; lanes 16-31: M=8+r, N=lane-16
    const int n_lo = lane & 15;
    const int hi   = lane >> 4;
#pragma unroll
    for (int i = 0; i < 2; ++i) {
        const int fr = row + i * 16;          // fragment row base (16-row span, one bias group)
        const int rbase = fr + hi * 8;
#pragma unroll
        for (int j = 0; j < 4; ++j) {
            const int n = col + j * 16 + n_lo;
            float add = 0.0f;
            if (biasN) add += biasN[n];
            if (biasG) add += biasG[(size_t)(fr >> gshift) * N + n];
#pragma unroll
            for (int r = 0; r < 8; ++r) {
                const float v = c[i][j][r] + add;
                const size_t off = (size_t)(rbase + r) * N + n;
                if (outB) outB[off] = (__bf16)v;
                else      outF[off] = v;
            }
        }
    }
}

// ---------------------------------------------------------------------------
// Fused attention step: scores = tanh(z1 + base)·W3 + b3 ; softmax over T ;
// ct = beta·h ; y_tilde = bt + Wt[0]*y_t + Wt[1:]·ct.  One block per batch row.
// ---------------------------------------------------------------------------
__global__ void score_ct_kernel(const __bf16* __restrict__ base, const __bf16* __restrict__ hb,
                                const float* __restrict__ z1, const float* __restrict__ W3,
                                const float* __restrict__ b3, const float* __restrict__ y_seq,
                                const float* __restrict__ Wt, const float* __restrict__ bt,
                                float* __restrict__ ytil, __bf16* __restrict__ dct,
                                int tstep, int last) {
    __shared__ float z1s[CH_DIM];
    __shared__ float w3s[CH_DIM];
    __shared__ float sc[T_DIM];
    __shared__ float cts[CH_DIM];
    __shared__ float red[256];

    const int b = blockIdx.x;
    const int tid = threadIdx.x;
    const int lane = tid & 31, wave = tid >> 5;

    for (int i = tid; i < CH_DIM; i += 256) { z1s[i] = z1[(size_t)b * CH_DIM + i]; w3s[i] = W3[i]; }
    __syncthreads();

    const __bf16* baseb = base + (size_t)b * T_DIM * CH_DIM;
    for (int t = wave; t < T_DIM; t += 8) {
        const __bf16* rowp = baseb + (size_t)t * CH_DIM;
        float acc = 0.0f;
        for (int cc = lane; cc < CH_DIM; cc += 32)
            acc += tanhf(z1s[cc] + (float)rowp[cc]) * w3s[cc];
        for (int off = 16; off; off >>= 1) acc += __shfl_down(acc, off, 32);
        if (lane == 0) sc[t] = acc + b3[0];
    }
    __syncthreads();

    if (wave == 0) {   // softmax over T=64 (2 values per lane)
        float s0 = sc[lane], s1 = sc[lane + 32];
        float m = fmaxf(s0, s1);
        for (int off = 16; off; off >>= 1) m = fmaxf(m, __shfl_down(m, off, 32));
        m = __shfl(m, 0, 32);
        float e0 = __expf(s0 - m), e1 = __expf(s1 - m);
        float ssum = e0 + e1;
        for (int off = 16; off; off >>= 1) ssum += __shfl_down(ssum, off, 32);
        ssum = __shfl(ssum, 0, 32);
        const float inv = 1.0f / ssum;
        sc[lane] = e0 * inv; sc[lane + 32] = e1 * inv;
    }
    __syncthreads();

    const __bf16* hbb = hb + (size_t)b * T_DIM * CH_DIM;
    float part = 0.0f;
    for (int cc = tid; cc < CH_DIM; cc += 256) {
        float acc = 0.0f;
#pragma unroll 8
        for (int t = 0; t < T_DIM; ++t) acc += sc[t] * (float)hbb[(size_t)t * CH_DIM + cc];
        cts[cc] = acc;
        part += acc * Wt[1 + cc];
        if (last) dct[(size_t)b * 1024 + CH_DIM + cc] = (__bf16)acc;
    }
    red[tid] = part;
    __syncthreads();
    for (int s = 128; s > 0; s >>= 1) {
        if (tid < s) red[tid] += red[tid + s];
        __syncthreads();
    }
    if (tid == 0) ytil[b] = bt[0] + Wt[0] * y_seq[(size_t)b * T_DIM + tstep] + red[0];
}

// ---------------------------------------------------------------------------
// LSTM cell update: gates already hold d@Whh^T + (bih+bhh); add y_tilde*Wih here.
// ---------------------------------------------------------------------------
__global__ void lstm_kernel(const float* __restrict__ gates, const float* __restrict__ ytil,
                            const float* __restrict__ Wih, float* __restrict__ s_f32,
                            __bf16* __restrict__ ds, __bf16* __restrict__ dct, int last) {
    const int idx = blockIdx.x * blockDim.x + threadIdx.x;  // B*H
    const int b = idx >> 9, j = idx & 511;
    const float yt = ytil[b];
    const float* g = gates + (size_t)b * (4 * H_DIM);
    const float gi = g[j]                + yt * Wih[j];
    const float gf = g[H_DIM + j]        + yt * Wih[H_DIM + j];
    const float gg = g[2 * H_DIM + j]    + yt * Wih[2 * H_DIM + j];
    const float go = g[3 * H_DIM + j]    + yt * Wih[3 * H_DIM + j];
    const float so = s_f32[idx];
    const float sn = sigm(gf) * so + sigm(gi) * tanhf(gg);
    const float dn = sigm(go) * tanhf(sn);
    s_f32[idx] = sn;
    ds[(size_t)b * (2 * H_DIM) + j]          = (__bf16)dn;
    ds[(size_t)b * (2 * H_DIM) + H_DIM + j]  = (__bf16)sn;
    if (last) dct[(size_t)b * 1024 + j] = (__bf16)dn;
}

// ---------------------------------------------------------------------------
// Host-side orchestration
// ---------------------------------------------------------------------------
static inline void launch_gemm(const __bf16* A, int lda, const __bf16* W, int ldw,
                               float* outF, __bf16* outB, const float* biasN,
                               const float* biasG, int gshift,
                               int M, int N, int K, hipStream_t s) {
    dim3 grid(N / 64, M / 256);
    gemm_wmma_kernel<<<grid, 256, 0, s>>>(A, lda, W, ldw, outF, outB, biasN, biasG, gshift, N, K);
}

static inline void launch_cvt(const float* in, __bf16* out, size_t n, hipStream_t s) {
    cvt_bf16_kernel<<<(unsigned)((n + 255) / 256), 256, 0, s>>>(in, out, n);
}

extern "C" void kernel_launch(void* const* d_in, const int* in_sizes, int n_in,
                              void* d_out, int out_size, void* d_ws, size_t ws_size,
                              hipStream_t stream) {
    (void)in_sizes; (void)n_in; (void)out_size; (void)ws_size;
    const float* h     = (const float*)d_in[0];
    const float* y_seq = (const float*)d_in[1];
    const float* cis   = (const float*)d_in[2];
    const float* W1    = (const float*)d_in[3];
    const float* b1    = (const float*)d_in[4];
    const float* W2    = (const float*)d_in[5];
    const float* b2    = (const float*)d_in[6];
    const float* Wci   = (const float*)d_in[7];
    const float* bci   = (const float*)d_in[8];
    const float* W3    = (const float*)d_in[9];
    const float* b3    = (const float*)d_in[10];
    const float* Wih   = (const float*)d_in[11];
    const float* Whh   = (const float*)d_in[12];
    const float* bih   = (const float*)d_in[13];
    const float* bhh   = (const float*)d_in[14];
    const float* Wt    = (const float*)d_in[15];
    const float* bt    = (const float*)d_in[16];
    const float* Wo    = (const float*)d_in[17];
    const float* bo    = (const float*)d_in[18];

    // Workspace carve-out (256B aligned)
    char* p = (char*)d_ws;
    auto carve = [&](size_t bytes) -> void* {
        void* r = (void*)p;
        p += (bytes + 255) & ~(size_t)255;
        return r;
    };
    const size_t hN = (size_t)B_DIM * T_DIM * CH_DIM;             // 33.5M
    __bf16* h_bf    = (__bf16*)carve(hN * 2);                     // 64 MB
    __bf16* base_bf = (__bf16*)carve(hN * 2);                     // 64 MB
    __bf16* W1b     = (__bf16*)carve((size_t)CH_DIM * 2 * H_DIM * 2);
    __bf16* W2b     = (__bf16*)carve((size_t)CH_DIM * CH_DIM * 2);
    __bf16* Whhb    = (__bf16*)carve((size_t)4 * H_DIM * H_DIM * 2);
    __bf16* Wob     = (__bf16*)carve((size_t)H_DIM * (CH_DIM + H_DIM) * 2);
    float*  zci     = (float*)carve((size_t)B_DIM * CH_DIM * 4);
    float*  bvec    = (float*)carve((size_t)4 * H_DIM * 4);
    __bf16* ds_bf   = (__bf16*)carve((size_t)B_DIM * 2 * H_DIM * 2);
    float*  s_f32   = (float*)carve((size_t)B_DIM * H_DIM * 4);
    float*  z1      = (float*)carve((size_t)B_DIM * CH_DIM * 4);
    float*  gates   = (float*)carve((size_t)B_DIM * 4 * H_DIM * 4);
    float*  ytil    = (float*)carve((size_t)B_DIM * 4);
    __bf16* dct_bf  = (__bf16*)carve((size_t)B_DIM * (CH_DIM + H_DIM) * 2);
    float*  outF    = (float*)d_out;

    // 1) precision conversions (every call: deterministic, no caching)
    launch_cvt(h,   h_bf, hN, stream);
    launch_cvt(W1,  W1b,  (size_t)CH_DIM * 2 * H_DIM, stream);
    launch_cvt(W2,  W2b,  (size_t)CH_DIM * CH_DIM, stream);
    launch_cvt(Whh, Whhb, (size_t)4 * H_DIM * H_DIM, stream);
    launch_cvt(Wo,  Wob,  (size_t)H_DIM * (CH_DIM + H_DIM), stream);

    // 2) loop-invariant terms
    zci_kernel<<<(B_DIM * CH_DIM) / 256, 256, 0, stream>>>(cis, Wci, bci, b2, zci);
    bvec_kernel<<<(4 * H_DIM) / 256, 256, 0, stream>>>(bih, bhh, bvec);

    // 3) base = h @ W2^T + (b2 + zci)   (grouped bias: row>>6 == batch index), bf16 out
    launch_gemm(h_bf, CH_DIM, W2b, CH_DIM, nullptr, base_bf,
                nullptr, zci, 6, B_DIM * T_DIM, CH_DIM, CH_DIM, stream);

    // 4) zero initial state
    hipMemsetAsync(ds_bf, 0, (size_t)B_DIM * 2 * H_DIM * 2, stream);
    hipMemsetAsync(s_f32, 0, (size_t)B_DIM * H_DIM * 4, stream);

    // 5) sequential scan over T steps
    for (int t = 0; t < T_DIM; ++t) {
        const int last = (t == T_DIM - 1);
        // z1 = [d,s] @ W1^T + b1       (M=B, K=2H, N=CH)
        launch_gemm(ds_bf, 2 * H_DIM, W1b, 2 * H_DIM, z1, nullptr,
                    b1, nullptr, 0, B_DIM, CH_DIM, 2 * H_DIM, stream);
        // gates = d @ Whh^T + (bih+bhh)   (M=B, K=H via lda=2H, N=4H) — independent of y_tilde
        launch_gemm(ds_bf, 2 * H_DIM, Whhb, H_DIM, gates, nullptr,
                    bvec, nullptr, 0, B_DIM, 4 * H_DIM, H_DIM, stream);
        // attention: scores -> softmax -> ct -> y_tilde
        score_ct_kernel<<<B_DIM, 256, 0, stream>>>(base_bf, h_bf, z1, W3, b3, y_seq,
                                                   Wt, bt, ytil, dct_bf, t, last);
        // LSTM update (adds y_tilde * Wih to gates)
        lstm_kernel<<<(B_DIM * H_DIM) / 256, 256, 0, stream>>>(gates, ytil, Wih,
                                                               s_f32, ds_bf, dct_bf, last);
    }

    // 6) out = [d, ct] @ Wo^T + bo     (M=B, K=CH+H, N=H) -> f32 d_out
    launch_gemm(dct_bf, CH_DIM + H_DIM, Wob, CH_DIM + H_DIM, outF, nullptr,
                bo, nullptr, 0, B_DIM, H_DIM, CH_DIM + H_DIM, stream);
}
